// ProtoNetGeneral_23270132809993
// MI455X (gfx1250) — matrix operand order repaired and checked by reading, not compiled, not run on hardware
//
#include <hip/hip_runtime.h>
#include <hip/hip_bf16.h>

typedef __attribute__((ext_vector_type(16))) __bf16 v16bf;
typedef __attribute__((ext_vector_type(8)))  __bf16 v8bf;
typedef __attribute__((ext_vector_type(8)))  float  v8f;

#define N_WAYS  512
#define N_ADAPT 51200
#define N_EVAL  65536
#define D_FEAT  512
#define MTILE   32
#define SK      520          // padded bf16 K-stride in LDS (bank-conflict free b128)
#define MLIST_CAP 8192

// ---------------------------------------------------------------------------
// Kernel A: per-class prototype (mean), bf16 hi/lo split, |p|^2, empty flag.
// One block per class, 512 threads (thread t owns feature t). Deterministic.
// ---------------------------------------------------------------------------
__global__ __launch_bounds__(512)
void proto_build(const float* __restrict__ xa, const int* __restrict__ ya,
                 __bf16* __restrict__ phi, __bf16* __restrict__ plo,
                 float* __restrict__ psq, int* __restrict__ inval) {
    const int cls = blockIdx.x;
    const int t   = threadIdx.x;              // 0..511
    __shared__ int   lcnt[512];
    __shared__ int   mlist[MLIST_CAP];
    __shared__ float red[512];

    // pass 1: count matches in this thread's contiguous slice (ordered)
    const int sbeg = t * (N_ADAPT / 512);
    const int send = sbeg + (N_ADAPT / 512);
    int c = 0;
    for (int s = sbeg; s < send; ++s) c += (ya[s] == cls) ? 1 : 0;
    lcnt[t] = c;
    __syncthreads();
    // inclusive Hillis-Steele scan over 512 counts
    for (int off = 1; off < 512; off <<= 1) {
        int add = (t >= off) ? lcnt[t - off] : 0;
        __syncthreads();
        lcnt[t] += add;
        __syncthreads();
    }
    int pos   = lcnt[t] - c;     // exclusive prefix
    int total = lcnt[511];
    // pass 2: fill match list in globally ascending sample order
    for (int s = sbeg; s < send; ++s) {
        if (ya[s] == cls) { if (pos < MLIST_CAP) mlist[pos] = s; ++pos; }
    }
    __syncthreads();
    int cnt = total < MLIST_CAP ? total : MLIST_CAP;

    float sum = 0.f;
    for (int i = 0; i < cnt; ++i)
        sum += xa[(size_t)mlist[i] * D_FEAT + t];
    float mean = sum / fmaxf((float)cnt, 1.0f);

    // bf16 split: mean = hi + lo (Markidis)
    __bf16 h = (__bf16)mean;
    float  hf = (float)h;
    __bf16 l = (__bf16)(mean - hf);
    phi[(size_t)cls * D_FEAT + t] = h;
    plo[(size_t)cls * D_FEAT + t] = l;

    red[t] = mean * mean;
    __syncthreads();
    for (int s = 256; s > 0; s >>= 1) {
        if (t < s) red[t] += red[t + s];
        __syncthreads();
    }
    if (t == 0) { psq[cls] = red[0]; inval[cls] = (total == 0) ? 1 : 0; }
}

// ---------------------------------------------------------------------------
// Kernel B: logits = 2*(q . p) - |q|^2 - |p|^2 via bf16x3 WMMA.
// 256 threads = 8 waves; M-tile = 32 rows in LDS (hi/lo bf16, padded stride);
// each wave owns 64 of the 512 classes: 2x4 tiles of 16x16 accumulators.
// ---------------------------------------------------------------------------
__global__ __launch_bounds__(256)
void gemm_logits(const float* __restrict__ xe,
                 const __bf16* __restrict__ phi, const __bf16* __restrict__ plo,
                 const float* __restrict__ psq, const int* __restrict__ inval,
                 float* __restrict__ out) {
    extern __shared__ char smem[];
    __bf16* ahi = (__bf16*)smem;                       // [MTILE][SK]
    __bf16* alo = (__bf16*)(smem + MTILE * SK * 2);    // [MTILE][SK]
    float*  qsq = (float*)(smem + 2 * MTILE * SK * 2); // [MTILE]

    const int tid = threadIdx.x;
    const int m0  = blockIdx.x * MTILE;

    if (tid < MTILE) qsq[tid] = 0.f;
    __syncthreads();

    // ---- Stage A: load 32x512 f32 tile, split to bf16 hi/lo in LDS, |q|^2 ----
    {
        const int row = tid >> 3;            // 0..31
        const int c0  = (tid & 7) * 64;      // 64 contiguous floats per thread
        const float4* src = (const float4*)(xe + (size_t)(m0 + row) * D_FEAT + c0);
        float q = 0.f;
        #pragma unroll
        for (int i = 0; i < 16; ++i) {
            float4 v = src[i];
            float a4[4] = {v.x, v.y, v.z, v.w};
            #pragma unroll
            for (int j = 0; j < 4; ++j) {
                float f  = a4[j];
                __bf16 h = (__bf16)f;
                float hf = (float)h;
                int idx  = row * SK + c0 + 4 * i + j;
                ahi[idx] = h;
                alo[idx] = (__bf16)(f - hf);
                q += f * f;
            }
        }
        atomicAdd(&qsq[row], q);             // ds_add_f32
    }
    __syncthreads();

    // ---- Stage B: WMMA K-loop ----
    const int lane = tid & 31;
    const int wave = tid >> 5;
    const int hl   = (lane >> 4) & 1;        // lane half
    const int l15  = lane & 15;
    const int nb   = wave * 64;              // this wave's class base

    v8f acc[2][4] = {};

    for (int k0 = 0; k0 < D_FEAT; k0 += 32) {
        // A fragments from LDS (contiguous K order, same as B):
        // lane<16: K runs k0..k0+15; lane>=16: shifted by 16.
        v16bf a_h[2], a_l[2];
        #pragma unroll
        for (int m = 0; m < 2; ++m) {
            int off = (m * 16 + l15) * SK + k0 + hl * 16;
            v8bf h0 = *(const v8bf*)(ahi + off);
            v8bf h1 = *(const v8bf*)(ahi + off + 8);
            v8bf l0 = *(const v8bf*)(alo + off);
            v8bf l1 = *(const v8bf*)(alo + off + 8);
            #pragma unroll
            for (int e = 0; e < 8; ++e) {
                a_h[m][e] = h0[e]; a_h[m][8 + e] = h1[e];
                a_l[m][e] = l0[e]; a_l[m][8 + e] = l1[e];
            }
        }
        #pragma unroll
        for (int n = 0; n < 4; ++n) {
            // B fragment: element e <-> K = k0 + 16*hl + e, N = l15 (proto is [N][K])
            size_t boff = (size_t)(nb + n * 16 + l15) * D_FEAT + k0 + hl * 16;
            v16bf bh = *(const v16bf*)(phi + boff);
            v16bf bl = *(const v16bf*)(plo + boff);
            #pragma unroll
            for (int m = 0; m < 2; ++m) {
                acc[m][n] = __builtin_amdgcn_wmma_f32_16x16x32_bf16(
                    false, a_h[m], false, bh, (short)0, acc[m][n], false, false);
                acc[m][n] = __builtin_amdgcn_wmma_f32_16x16x32_bf16(
                    false, a_h[m], false, bl, (short)0, acc[m][n], false, false);
                acc[m][n] = __builtin_amdgcn_wmma_f32_16x16x32_bf16(
                    false, a_l[m], false, bh, (short)0, acc[m][n], false, false);
            }
        }
    }

    // ---- Epilogue: logits = 2*acc - |q|^2 - |p|^2, invalid-class row fill ----
    #pragma unroll
    for (int m = 0; m < 2; ++m) {
        #pragma unroll
        for (int n = 0; n < 4; ++n) {
            int col  = nb + n * 16 + l15;
            float ps = psq[col];
            #pragma unroll
            for (int r = 0; r < 8; ++r) {
                int lrow = m * 16 + r + hl * 8;   // C layout: lanes>=16 hold M=8..15
                int grow = m0 + lrow;
                float val = 2.0f * acc[m][n][r] - qsq[lrow] - ps;
                if (grow < N_WAYS && inval[grow]) val = -1.0e6f;
                out[(size_t)grow * N_WAYS + col] = val;
            }
        }
    }
}

// ---------------------------------------------------------------------------
// Kernel C: softmax over 512 classes, first-occurrence argmax, top prob.
// One wave (32 lanes) per row; 8 rows per 256-thread block.
// ---------------------------------------------------------------------------
__global__ __launch_bounds__(256)
void softmax_top(const float* __restrict__ logits, float* __restrict__ ph,
                 float* __restrict__ yh, float* __restrict__ pt) {
    const int lane = threadIdx.x & 31;
    const int row  = blockIdx.x * 8 + (threadIdx.x >> 5);
    const float* lr = logits + (size_t)row * N_WAYS;

    float v[16];
    float bmax = -__builtin_inff();
    int   bcol = 0;
    #pragma unroll
    for (int j = 0; j < 16; ++j) {
        float x = lr[lane + 32 * j];
        v[j] = x;
        if (x > bmax) { bmax = x; bcol = lane + 32 * j; }
    }
    #pragma unroll
    for (int off = 16; off > 0; off >>= 1) {
        float ov = __shfl_xor(bmax, off, 32);
        int   oc = __shfl_xor(bcol, off, 32);
        if (ov > bmax || (ov == bmax && oc < bcol)) { bmax = ov; bcol = oc; }
    }
    float s = 0.f;
    #pragma unroll
    for (int j = 0; j < 16; ++j) { v[j] = __expf(v[j] - bmax); s += v[j]; }
    #pragma unroll
    for (int off = 16; off > 0; off >>= 1) s += __shfl_xor(s, off, 32);
    float inv = 1.0f / s;

    float* pr = ph + (size_t)row * N_WAYS;
    #pragma unroll
    for (int j = 0; j < 16; ++j) pr[lane + 32 * j] = v[j] * inv;
    if (lane == 0) { yh[row] = (float)bcol; pt[row] = inv; }  // max ph = 1/s
}

// ---------------------------------------------------------------------------
extern "C" void kernel_launch(void* const* d_in, const int* in_sizes, int n_in,
                              void* d_out, int out_size, void* d_ws, size_t ws_size,
                              hipStream_t stream) {
    const float* xa = (const float*)d_in[0];   // [51200, 512] f32
    const int*   ya = (const int*)d_in[1];     // [51200] labels
    const float* xe = (const float*)d_in[2];   // [65536, 512] f32

    char* ws = (char*)d_ws;
    __bf16* phi   = (__bf16*)ws;                                // 512*512*2 B
    __bf16* plo   = (__bf16*)(ws + (size_t)N_WAYS * D_FEAT * 2);
    float*  psq   = (float*)(ws + (size_t)N_WAYS * D_FEAT * 4);
    int*    inval = (int*)(ws + (size_t)N_WAYS * D_FEAT * 4 + 2048);

    float* out = (float*)d_out;                                  // logits
    float* ph  = out + (size_t)N_EVAL * N_WAYS;                  // softmax
    float* yh  = ph + (size_t)N_EVAL * N_WAYS;                   // argmax (as float)
    float* pt  = yh + N_EVAL;                                    // top prob

    proto_build<<<N_WAYS, 512, 0, stream>>>(xa, ya, phi, plo, psq, inval);

    const size_t lds = 2 * (size_t)MTILE * SK * 2 + MTILE * 4;   // hi + lo + qsq
    gemm_logits<<<N_EVAL / MTILE, 256, lds, stream>>>(xe, phi, plo, psq, inval, out);

    softmax_top<<<N_EVAL / 8, 256, 0, stream>>>(out, ph, yh, pt);
}
